// TriangleAttention_52407190946183
// MI455X (gfx1250) — matrix operand order repaired and checked by reading, not compiled, not run on hardware
//
#include <hip/hip_runtime.h>
#include <cstdint>
#include <cstddef>

// ---------------------------------------------------------------------------
// TriangleAttention on gfx1250 (MI455X), bf16 WMMA pipeline.
// B=2, N=256, C_IN=128, H=4, C_HID=32.
// ---------------------------------------------------------------------------

typedef __attribute__((ext_vector_type(16))) __bf16 v16bf;
typedef __attribute__((ext_vector_type(8)))  float  v8f;
typedef __attribute__((ext_vector_type(2)))  __bf16 bf16x2;

#define DEV static __device__ __forceinline__

struct U32x8 { uint4 a; uint4 b; };        // 32 bytes
struct BF4   { __bf16 a, b, c, d; };       // 8 bytes

DEV bf16x2 ld_bf2(const __bf16* p) {
  return __builtin_bit_cast(bf16x2, *reinterpret_cast<const uint32_t*>(p));
}

// Load a 32-byte contiguous run of 16 bf16 -> B fragment (lane-local slice).
DEV v16bf ld_bf16x16(const __bf16* p) {
  U32x8 r;
  r.a = *reinterpret_cast<const uint4*>(p);
  r.b = *reinterpret_cast<const uint4*>(p + 8);
  return __builtin_bit_cast(v16bf, r);
}

// A-matrix 16x32 bf16 fragment (per-lane slice) from a row-major bf16 row.
DEV v16bf load_afrag(const __bf16* rowp, int hi, int kbase) {
  v16bf a;
#pragma unroll
  for (int v = 0; v < 4; ++v) {
    bf16x2 p0 = ld_bf2(rowp + kbase + hi * 16 + 2 * v);
    a[2 * v]     = p0[0];
    a[2 * v + 1] = p0[1];
    bf16x2 p1 = ld_bf2(rowp + kbase + hi * 16 + 8 + 2 * v);
    a[8 + 2 * v]     = p1[0];
    a[8 + 2 * v + 1] = p1[1];
  }
  return a;
}

DEV v8f wmma_bf16(v16bf a, v16bf b, v8f c) {
  return __builtin_amdgcn_wmma_f32_16x16x32_bf16(
      false, a, false, b, (short)0, c, false, false);
}

DEV v8f v8f_zero() {
  v8f z = {0.f, 0.f, 0.f, 0.f, 0.f, 0.f, 0.f, 0.f};
  return z;
}

DEV float sigm(float x) { return 1.0f / (1.0f + __expf(-x)); }

// 16-byte global -> LDS async copy (ASYNCcnt-tracked, CDNA5 ISA 15.18.3).
// VDST holds the per-lane LDS byte offset; for a generic pointer into LDS the
// low 32 bits ARE the LDS address (aperture rules, ISA 10.2), so a simple
// truncation gives the operand without any address-space gymnastics.
DEV void async_cp16(const __bf16* g, __bf16* l) {
  uint32_t lofs = (uint32_t)(uintptr_t)l;
  asm volatile("global_load_async_to_lds_b128 %0, %1, off"
               :: "v"(lofs), "v"(g)
               : "memory");
}

DEV void wait_async_done() {
  asm volatile("s_wait_asynccnt 0x0" ::: "memory");
}

// ---------------------------------------------------------------------------
// Kernel 1: convert+transpose the 5 weight matrices (wq,wk,wv,wg,wo) to bf16.
// wT[m][n*128 + k] = w_m[k*128 + n]  (so B-fragments are contiguous in K).
// ---------------------------------------------------------------------------
__global__ void k_prep_w(const float* __restrict__ wq, const float* __restrict__ wk,
                         const float* __restrict__ wv, const float* __restrict__ wg,
                         const float* __restrict__ wo, __bf16* __restrict__ wT) {
  int idx = blockIdx.x * blockDim.x + threadIdx.x;
  if (idx >= 5 * 128 * 128) return;
  int m = idx >> 14;
  int rem = idx & 16383;
  int n = rem >> 7, k = rem & 127;
  const float* src = (m == 0) ? wq : (m == 1) ? wk : (m == 2) ? wv : (m == 3) ? wg : wo;
  wT[idx] = (__bf16)src[k * 128 + n];
}

// ---------------------------------------------------------------------------
// Kernel 2: LayerNorm (per 128-ch row) + triangle bias. One wave per row.
// ---------------------------------------------------------------------------
__global__ __launch_bounds__(256) void k_ln(const float* __restrict__ x,
                                            const float* __restrict__ lnw,
                                            const float* __restrict__ lnb,
                                            const float* __restrict__ wtri,
                                            __bf16* __restrict__ xn,
                                            float* __restrict__ tri) {
  int wid = threadIdx.x >> 5, lane = threadIdx.x & 31;
  long row = (long)blockIdx.x * 8 + wid;          // 0 .. 131071
  const float* xr = x + row * 128;
  float4 xv = *reinterpret_cast<const float4*>(xr + lane * 4);
  float s  = xv.x + xv.y + xv.z + xv.w;
  float s2 = xv.x * xv.x + xv.y * xv.y + xv.z * xv.z + xv.w * xv.w;
#pragma unroll
  for (int m = 16; m >= 1; m >>= 1) {
    s  += __shfl_xor(s,  m, 32);
    s2 += __shfl_xor(s2, m, 32);
  }
  float mu  = s * (1.0f / 128.0f);
  float var = s2 * (1.0f / 128.0f) - mu * mu;
  float inv = rsqrtf(var + 1e-5f);
  float4 wv4 = *reinterpret_cast<const float4*>(lnw + lane * 4);
  float4 bv4 = *reinterpret_cast<const float4*>(lnb + lane * 4);
  float n0 = (xv.x - mu) * inv * wv4.x + bv4.x;
  float n1 = (xv.y - mu) * inv * wv4.y + bv4.y;
  float n2 = (xv.z - mu) * inv * wv4.z + bv4.z;
  float n3 = (xv.w - mu) * inv * wv4.w + bv4.w;
  BF4 outw = {(__bf16)n0, (__bf16)n1, (__bf16)n2, (__bf16)n3};
  *reinterpret_cast<uint2*>(xn + row * 128 + lane * 4) =
      __builtin_bit_cast(uint2, outw);

  long b = row >> 16;
  long i = (row >> 8) & 255;
  long j = row & 255;
#pragma unroll
  for (int h = 0; h < 4; ++h) {
    float4 wt = *reinterpret_cast<const float4*>(wtri + h * 128 + lane * 4);
    float t = n0 * wt.x + n1 * wt.y + n2 * wt.z + n3 * wt.w;
#pragma unroll
    for (int m = 16; m >= 1; m >>= 1) t += __shfl_xor(t, m, 32);
    if (lane == 0) tri[((b * 4 + h) * 256 + i) * 256 + j] = t;
  }
}

// ---------------------------------------------------------------------------
// Kernel 3: q/k/v/g projections via WMMA. One wave per 16-row M-tile.
// q pre-scaled by 1/sqrt(32); bg folded into g; v stored transposed.
// ---------------------------------------------------------------------------
__global__ __launch_bounds__(256) void k_proj(const __bf16* __restrict__ xn,
                                              const __bf16* __restrict__ wT,
                                              const float* __restrict__ bg,
                                              __bf16* __restrict__ qb,
                                              __bf16* __restrict__ kb,
                                              __bf16* __restrict__ vT,
                                              __bf16* __restrict__ gb) {
  int wid = threadIdx.x >> 5, lane = threadIdx.x & 31;
  int hi = lane >> 4, ln = lane & 15;
  long tile = (long)blockIdx.x * 8 + wid;         // 8192 tiles of 16 rows
  long rowbase = tile * 16;
  __builtin_prefetch(wT, 0, 1);

  const __bf16* arow = xn + (rowbase + ln) * 128;
  v16bf a[4];
#pragma unroll
  for (int kc = 0; kc < 4; ++kc) a[kc] = load_afrag(arow, hi, kc * 32);

#pragma unroll
  for (int m = 0; m < 4; ++m) {
    const __bf16* wTm = wT + m * 16384;
    __bf16* outp = (m == 0) ? qb : (m == 1) ? kb : (m == 2) ? vT : gb;
    float scale = (m == 0) ? 0.17677669529663687f : 1.0f;   // 1/sqrt(32)
#pragma unroll
    for (int nt = 0; nt < 8; ++nt) {
      int col = nt * 16 + ln;
      v8f acc = v8f_zero();
#pragma unroll
      for (int kc = 0; kc < 4; ++kc) {
        v16bf bf = ld_bf16x16(wTm + col * 128 + kc * 32 + hi * 16);
        acc = wmma_bf16(a[kc], bf, acc);
      }
      float badd = (m == 3) ? bg[col] : 0.0f;
#pragma unroll
      for (int r = 0; r < 8; ++r) {
        float vv = acc[r] * scale + badd;
        long row = rowbase + r + hi * 8;
        if (m == 2) {
          long bi = row >> 8;
          long j  = row & 255;
          vT[(bi * 128 + col) * 256 + j] = (__bf16)vv;
        } else {
          outp[row * 128 + col] = (__bf16)vv;
        }
      }
    }
  }
}

// ---------------------------------------------------------------------------
// Kernel 4: attention. One block per (b,i,h): stage K (256x32) and V (32x256)
// tiles into LDS once via global_load_async_to_lds_b128, then 2 waves sweep
// the 16 q-tiles reading B fragments from LDS. Mask+triangle bias initializes
// the WMMA C accumulator; sigmoid gate fused into the epilogue.
// LDS: sK 256x40 (20.0KB) + sV 32x264 (16.5KB) + sP 2x16x264 (16.5KB) = 53KB.
// ---------------------------------------------------------------------------
__global__ __launch_bounds__(64) void k_att(const __bf16* __restrict__ qb,
                                            const __bf16* __restrict__ kb,
                                            const __bf16* __restrict__ vT,
                                            const float* __restrict__ tri,
                                            const float* __restrict__ mask,
                                            const __bf16* __restrict__ gb,
                                            __bf16* __restrict__ ob) {
  __shared__ __bf16 sK[256 * 40];        // [key][ch padded 32->40]
  __shared__ __bf16 sV[32 * 264];        // [ch][key padded 256->264]
  __shared__ __bf16 sP[2][16 * 264];     // per-wave probs [q][key padded]

  int tid = threadIdx.x;
  int wid = tid >> 5, lane = tid & 31;
  int hi = lane >> 4, ln = lane & 15;
  long id = blockIdx.x;                  // (b*N + i)*H + h
  int h = (int)(id & 3);
  long bi = id >> 2;                     // b*256 + i
  long b  = bi >> 8;

  // --- stage K (wave 0) and V (wave 1) tiles into LDS -----------------------
  if (wid == 0) {
    for (int it = 0; it < 32; ++it) {    // 8 keys per step, 4 lanes per key
      int key = it * 8 + (lane >> 2);
      int seg = lane & 3;                // 16B segment of the 64B key row
      const __bf16* g = kb + (bi * 256 + key) * 128 + h * 32 + seg * 8;
      async_cp16(g, &sK[key * 40 + seg * 8]);
    }
  } else {
    for (int ch = 0; ch < 32; ++ch) {    // one 512B channel row per step
      const __bf16* g = vT + (bi * 128 + h * 32 + ch) * 256 + lane * 8;
      async_cp16(g, &sV[ch * 264 + lane * 8]);
    }
  }
  wait_async_done();
  __syncthreads();

  const float* trip  = tri + (b * 4 + h) * 65536;
  const float* maskp = mask + bi * 256;

  for (int it = 0; it < 8; ++it) {
    int qt = it * 2 + wid;

    // Q fragment (16x32, K = head channels)
    const __bf16* qrow = qb + (bi * 256 + qt * 16 + ln) * 128 + h * 32;
    v16bf qa = load_afrag(qrow, hi, 0);

    // logits = Q K^T + (mask bias + triangle bias) as WMMA C input
    v8f lg[16];
#pragma unroll
    for (int kt = 0; kt < 16; ++kt) {
      int key = kt * 16 + ln;
      float mb = 1e9f * (maskp[key] - 1.0f);
      v8f cb;
#pragma unroll
      for (int r = 0; r < 8; ++r)
        cb[r] = mb + trip[(qt * 16 + r + hi * 8) * 256 + key];
      v16bf kf = ld_bf16x16(&sK[key * 40 + hi * 16]);
      lg[kt] = wmma_bf16(qa, kf, cb);
    }

    // softmax over 256 keys per query row
#pragma unroll
    for (int r = 0; r < 8; ++r) {
      float mx = -3.0e38f;
#pragma unroll
      for (int kt = 0; kt < 16; ++kt) mx = fmaxf(mx, lg[kt][r]);
#pragma unroll
      for (int m = 8; m >= 1; m >>= 1) mx = fmaxf(mx, __shfl_xor(mx, m, 32));
      float sum = 0.f;
#pragma unroll
      for (int kt = 0; kt < 16; ++kt) {
        float e = __expf(lg[kt][r] - mx);
        lg[kt][r] = e;
        sum += e;
      }
#pragma unroll
      for (int m = 8; m >= 1; m >>= 1) sum += __shfl_xor(sum, m, 32);
      float inv = 1.0f / sum;
#pragma unroll
      for (int kt = 0; kt < 16; ++kt) lg[kt][r] *= inv;
    }

    // stage probs (C layout -> row-major bf16), reload as A fragments
    __bf16* myP = &sP[wid][0];
#pragma unroll
    for (int kt = 0; kt < 16; ++kt)
#pragma unroll
      for (int r = 0; r < 8; ++r)
        myP[(r + hi * 8) * 264 + kt * 16 + ln] = (__bf16)lg[kt][r];

    // O = P @ V   (K = 256 keys in 8 chunks; N = 32 channels -> 2 tiles)
    v8f o0 = v8f_zero(), o1 = v8f_zero();
#pragma unroll
    for (int kc = 0; kc < 8; ++kc) {
      v16bf pa = load_afrag(myP + ln * 264, hi, kc * 32);
      v16bf v0 = ld_bf16x16(&sV[ln * 264 + kc * 32 + hi * 16]);
      o0 = wmma_bf16(pa, v0, o0);
      v16bf v1 = ld_bf16x16(&sV[(16 + ln) * 264 + kc * 32 + hi * 16]);
      o1 = wmma_bf16(pa, v1, o1);
    }

    // epilogue: fuse sigmoid gate, store bf16
#pragma unroll
    for (int r = 0; r < 8; ++r) {
      long orow = bi * 256 + qt * 16 + r + hi * 8;
      int c0 = h * 32 + ln, c1 = c0 + 16;
      float g0 = sigm((float)gb[orow * 128 + c0]);
      float g1 = sigm((float)gb[orow * 128 + c1]);
      ob[orow * 128 + c0] = (__bf16)(o0[r] * g0);
      ob[orow * 128 + c1] = (__bf16)(o1[r] * g1);
    }
  }
}

// ---------------------------------------------------------------------------
// Kernel 5: output projection (wo) + bo on the pre-gated ob. f32 output.
// ---------------------------------------------------------------------------
__global__ __launch_bounds__(256) void k_out(const __bf16* __restrict__ ob,
                                             const __bf16* __restrict__ wT,
                                             const float* __restrict__ bo,
                                             float* __restrict__ out) {
  int wid = threadIdx.x >> 5, lane = threadIdx.x & 31;
  int hi = lane >> 4, ln = lane & 15;
  long tile = (long)blockIdx.x * 8 + wid;
  long rowbase = tile * 16;
  const __bf16* orow = ob + (rowbase + ln) * 128;

  v16bf a[4];
#pragma unroll
  for (int kc = 0; kc < 4; ++kc) a[kc] = load_afrag(orow, hi, kc * 32);

  const __bf16* woT = wT + 4 * 16384;
#pragma unroll
  for (int nt = 0; nt < 8; ++nt) {
    int col = nt * 16 + ln;
    v8f acc = v8f_zero();
#pragma unroll
    for (int kc = 0; kc < 4; ++kc) {
      v16bf bf = ld_bf16x16(woT + col * 128 + kc * 32 + hi * 16);
      acc = wmma_bf16(a[kc], bf, acc);
    }
    float bv = bo[col];
#pragma unroll
    for (int r = 0; r < 8; ++r)
      out[(rowbase + r + hi * 8) * 128 + col] = acc[r] + bv;
  }
}

// ---------------------------------------------------------------------------
// Host-side launch. Inputs (setup_inputs order):
// 0:x 1:mask 2:ln_w 3:ln_b 4:w_tri 5:wq 6:wk 7:wv 8:wg 9:bg 10:wo 11:bo
// ---------------------------------------------------------------------------
extern "C" void kernel_launch(void* const* d_in, const int* in_sizes, int n_in,
                              void* d_out, int out_size, void* d_ws, size_t ws_size,
                              hipStream_t stream) {
  const float* x    = (const float*)d_in[0];
  const float* mask = (const float*)d_in[1];
  const float* lnw  = (const float*)d_in[2];
  const float* lnb  = (const float*)d_in[3];
  const float* wtri = (const float*)d_in[4];
  const float* wq   = (const float*)d_in[5];
  const float* wk   = (const float*)d_in[6];
  const float* wv   = (const float*)d_in[7];
  const float* wg   = (const float*)d_in[8];
  const float* bg   = (const float*)d_in[9];
  const float* wo   = (const float*)d_in[10];
  const float* bo   = (const float*)d_in[11];

  uint8_t* ws = (uint8_t*)d_ws;
  const size_t SZ_ACT = 2ull * 256 * 256 * 128 * 2;   // bf16 activation tensor
  size_t off = 0;
  __bf16* xn  = (__bf16*)(ws + off); off += SZ_ACT;
  float*  tri = (float*)(ws + off);  off += 2ull * 4 * 256 * 256 * 4;
  __bf16* qb  = (__bf16*)(ws + off); off += SZ_ACT;
  __bf16* kb  = (__bf16*)(ws + off); off += SZ_ACT;
  __bf16* vT  = (__bf16*)(ws + off); off += SZ_ACT;
  __bf16* gb  = (__bf16*)(ws + off); off += SZ_ACT;
  __bf16* ob  = (__bf16*)(ws + off); off += SZ_ACT;
  __bf16* wT  = (__bf16*)(ws + off); off += 5ull * 128 * 128 * 2;

  k_prep_w<<<(5 * 16384 + 255) / 256, 256, 0, stream>>>(wq, wk, wv, wg, wo, wT);
  k_ln<<<16384, 256, 0, stream>>>(x, lnw, lnb, wtri, xn, tri);
  k_proj<<<1024, 256, 0, stream>>>(xn, wT, bg, qb, kb, vT, gb);
  k_att<<<2048, 64, 0, stream>>>(qb, kb, vT, tri, mask, gb, ob);
  k_out<<<1024, 256, 0, stream>>>(ob, wT, bo, (float*)d_out);
}
